// TitansMemoryModule_18537078849838
// MI455X (gfx1250) — matrix operand order repaired and checked by reading, not compiled, not run on hardware
//
#include <hip/hip_runtime.h>
#include <hip/hip_bf16.h>

// ---------------------------------------------------------------------------
// Titans memory-module update for MI455X (gfx1250), wave32 + WMMA f16.
// B=65536, KD=VD=512.  d_out = [retrieved(B*VD) | loss(1) | new_W(VD*KD) | new_S(VD*KD)]
//
// Roofline: min traffic = k(128MB) + v(128MB) + retrieved(128MB) + ~4MB
// ~= 390 MB -> ~16.7 us @ 23.3 TB/s. 68.7 GFLOP in f16 WMMA hides under that.
// k band kept resident in LDS (f16) so it is read from HBM exactly once.
// ---------------------------------------------------------------------------

#define BDIM 65536
#define KD   512
#define VD   512

typedef __attribute__((ext_vector_type(16))) _Float16 v16h;
typedef __attribute__((ext_vector_type(8)))  float    v8f;

// ----- WMMA fragment loaders (contiguous k order: element s holds k = 16*(lane/16) + s, wave32) -----------

// A (16x32, f16), tile row-major [m][k]: lane L -> M = L&15; lanes<16 hold
// K=0..15, lanes>=16 hold K=16..31, 2 halfs per VGPR.
__device__ inline v16h load_a_rm(const _Float16* lds, int m0, int k0, int stride, int lane) {
    const int M  = lane & 15;
    const int kb = (lane >> 4) << 4;          // 0 or 16
    v16h a;
#pragma unroll
    for (int vv = 0; vv < 8; ++vv) {
        const int kl = kb + 2 * vv;
        const _Float16* p = lds + (m0 + M) * stride + k0 + kl;
        a[2 * vv]     = p[0];
        a[2 * vv + 1] = p[1];
    }
    return a;
}

// B (32x16, f16), tile row-major [k][n]: lane L -> N = L&15; lanes<16 hold
// K=0..15, lanes>=16 hold K=16..31, 2 halfs per VGPR.
__device__ inline v16h load_b_rm(const _Float16* lds, int k0, int n0, int stride, int lane) {
    const int N  = lane & 15;
    const int kb = (lane >> 4) << 4;          // 0 or 16
    v16h b;
#pragma unroll
    for (int vv = 0; vv < 8; ++vv) {
        b[2 * vv]     = lds[(k0 + kb + 2 * vv)     * stride + n0 + N];
        b[2 * vv + 1] = lds[(k0 + kb + 2 * vv + 1) * stride + n0 + N];
    }
    return b;
}

// B tile stored transposed in LDS (W tile is [n][k] row-major): B[k][n] = lds[n][k]
__device__ inline v16h load_b_tr(const _Float16* lds, int k0, int n0, int stride, int lane) {
    const int N  = lane & 15;
    const int kb = (lane >> 4) << 4;
    const _Float16* p = lds + (n0 + N) * stride + k0 + kb;
    v16h b;
#pragma unroll
    for (int vv = 0; vv < 8; ++vv) {
        b[2 * vv]     = p[2 * vv];
        b[2 * vv + 1] = p[2 * vv + 1];
    }
    return b;
}

// ---------------------------------------------------------------------------
__global__ void zero_ws_kernel(float* ws, int n) {
    int i = blockIdx.x * blockDim.x + threadIdx.x;
    if (i < n) ws[i] = 0.0f;
}

// 1 wave32: gates[g] = sigmoid(gate_w[g] . mean_k + gate_b[g])
__global__ void gates_kernel(const float* __restrict__ ksum,
                             const float* __restrict__ gw,
                             const float* __restrict__ gb,
                             float* __restrict__ gates) {
    const int lane = threadIdx.x;  // 32 threads
#pragma unroll
    for (int g = 0; g < 3; ++g) {
        float p = 0.0f;
        for (int j = lane; j < KD; j += 32) p += gw[g * KD + j] * ksum[j];
#pragma unroll
        for (int off = 16; off > 0; off >>= 1) p += __shfl_xor(p, off, 32);
        if (lane == 0) {
            float x = p * (1.0f / (float)BDIM) + gb[g];
            gates[g] = 1.0f / (1.0f + __expf(-x));
        }
    }
}

// ---------------------------------------------------------------------------
// Fused: retrieved = k@W.T, diff, loss partial, ksum partial, grad += diff.T@k.
// Block: 256 threads = 8 waves.  Tile: 128 rows x 64 cols.
// k band (128 x 512) kept resident in LDS as f16 -> k read from HBM once.
// ---------------------------------------------------------------------------
#define STKB 520  // resident k band LDS stride (halfs): 128 x 512 payload
#define STW  40   // W tile LDS stride (halfs): 64 x 32 payload
#define STD  136  // diff tile LDS stride (halfs), stored TRANSPOSED: 64 x 128 payload

__global__ __launch_bounds__(256)
void titans_main_kernel(const float* __restrict__ k, const float* __restrict__ v,
                        const float* __restrict__ W, float* __restrict__ retrieved,
                        float* __restrict__ gradAcc, float* __restrict__ lossAcc,
                        float* __restrict__ ksum) {
    __shared__ _Float16 kLDS[128 * STKB];   // 133,120 B
    __shared__ _Float16 wT[64 * STW];       //   5,120 B
    __shared__ _Float16 dT[64 * STD];       //  17,408 B (dT[vd][row])

    const int tid  = threadIdx.x;
    const int wave = tid >> 5;
    const int lane = tid & 31;
    const int row0 = blockIdx.x * 128;
    const int col0 = blockIdx.y * 64;
    const int wm = wave & 3;       // wave's 32-row sub-band (0..3)
    const int wn = wave >> 2;      // wave's 32-col sub-band (0..1)

    v8f c[2][2] = {};

    // ---- GEMM: retrieved tile = k(band) @ W(band).T, K in chunks of 32 ----
    for (int kk = 0; kk < KD; kk += 32) {
        {   // k chunk: 128x32 fp32 -> f16 into resident kLDS[:, kk:kk+32]
            const int r  = tid >> 1;
            const int cb = (tid & 1) * 16;
            const float* src = k + (size_t)(row0 + r) * KD + kk + cb;
            if (kk + 32 < KD) __builtin_prefetch(src + 32, 0, 3);
            _Float16* dst = kLDS + r * STKB + kk + cb;
#pragma unroll
            for (int j = 0; j < 16; j += 4) {
                float4 f = *reinterpret_cast<const float4*>(src + j);
                dst[j + 0] = (_Float16)f.x; dst[j + 1] = (_Float16)f.y;
                dst[j + 2] = (_Float16)f.z; dst[j + 3] = (_Float16)f.w;
            }
        }
        {   // W tile: 64x32; 4 threads per row, 8 floats each
            const int r  = tid >> 2;
            const int cb = (tid & 3) * 8;
            const float* src = W + (size_t)(col0 + r) * KD + kk + cb;
            _Float16* dst = wT + r * STW + cb;
#pragma unroll
            for (int j = 0; j < 8; j += 4) {
                float4 f = *reinterpret_cast<const float4*>(src + j);
                dst[j + 0] = (_Float16)f.x; dst[j + 1] = (_Float16)f.y;
                dst[j + 2] = (_Float16)f.z; dst[j + 3] = (_Float16)f.w;
            }
        }
        __syncthreads();

        v16h a0 = load_a_rm(kLDS, wm * 32,      kk, STKB, lane);
        v16h a1 = load_a_rm(kLDS, wm * 32 + 16, kk, STKB, lane);
        v16h b0 = load_b_tr(wT, 0, wn * 32,      STW, lane);
        v16h b1 = load_b_tr(wT, 0, wn * 32 + 16, STW, lane);
        c[0][0] = __builtin_amdgcn_wmma_f32_16x16x32_f16(false, a0, false, b0, (short)0, c[0][0], false, false);
        c[0][1] = __builtin_amdgcn_wmma_f32_16x16x32_f16(false, a0, false, b1, (short)0, c[0][1], false, false);
        c[1][0] = __builtin_amdgcn_wmma_f32_16x16x32_f16(false, a1, false, b0, (short)0, c[1][0], false, false);
        c[1][1] = __builtin_amdgcn_wmma_f32_16x16x32_f16(false, a1, false, b1, (short)0, c[1][1], false, false);
        __syncthreads();
    }

    // ---- Epilogue: store retrieved, diff -> LDS (transposed, f16), loss ---
    float ls = 0.0f;
    const int rowoff = (lane >> 4) * 8;
    const int nl = lane & 15;
#pragma unroll
    for (int mi = 0; mi < 2; ++mi) {
#pragma unroll
        for (int ni = 0; ni < 2; ++ni) {
            const int colL = wn * 32 + ni * 16 + nl;
            const int col  = col0 + colL;
#pragma unroll
            for (int r = 0; r < 8; ++r) {
                const int rowL = wm * 32 + mi * 16 + rowoff + r;
                const int row  = row0 + rowL;
                const float rv = c[mi][ni][r];
                retrieved[(size_t)row * VD + col] = rv;
                const float d = rv - v[(size_t)row * VD + col];
                ls += d * d;
                dT[colL * STD + rowL] = (_Float16)d;   // transposed: dT[vd][row]
            }
        }
    }
#pragma unroll
    for (int off = 16; off > 0; off >>= 1) ls += __shfl_xor(ls, off, 32);
    if (lane == 0) atomicAdd(lossAcc, ls);
    __syncthreads();

    // ---- Column sums of k band (for gates), from resident f16 LDS --------
    if (blockIdx.y == 0) {
        const int c0 = tid;          // 0..255
        const int c1 = tid + 256;    // 256..511
        float s0 = 0.0f, s1 = 0.0f;
#pragma unroll 4
        for (int r = 0; r < 128; ++r) {
            s0 += (float)kLDS[r * STKB + c0];
            s1 += (float)kLDS[r * STKB + c1];
        }
        atomicAdd(&ksum[c0], s0);
        atomicAdd(&ksum[c1], s1);
    }

    // ---- Grad pass: grad[col0:+64, :] += diff(band).T @ k(band) ----------
    // All operands are read-only in LDS now; no barriers needed inside.
    const int gm0 = (wave & 3) * 16;    // vd sub-tile within 64
    const int gn0 = (wave >> 2) * 16;   // kd sub-tile within 32-chunk
    for (int kc = 0; kc < KD; kc += 32) {
        v8f g = {};
#pragma unroll
        for (int rc = 0; rc < 128; rc += 32) {
            v16h at = load_a_rm(dT, gm0, rc, STD, lane);          // A = diff.T (M=vd, K=row)
            v16h bk = load_b_rm(kLDS, rc, kc + gn0, STKB, lane);  // B = k      (K=row, N=kd)
            g = __builtin_amdgcn_wmma_f32_16x16x32_f16(false, at, false, bk, (short)0, g, false, false);
        }
#pragma unroll
        for (int r = 0; r < 8; ++r) {
            const int vd = col0 + gm0 + rowoff + r;
            const int kd = kc + gn0 + nl;
            atomicAdd(&gradAcc[(size_t)vd * KD + kd], g[r]);
        }
    }
}

// ---------------------------------------------------------------------------
// new_S = eta*S - theta*(c*grad);  new_W = (1-alpha)*W + new_S;  loss finalize
// ---------------------------------------------------------------------------
__global__ __launch_bounds__(256)
void epilogue_kernel(const float* __restrict__ W, const float* __restrict__ S,
                     const float* __restrict__ gradAcc, const float* __restrict__ gates,
                     const float* __restrict__ lossAcc, float* __restrict__ out) {
    const int i = blockIdx.x * blockDim.x + threadIdx.x;   // 0 .. 262143
    const float alpha = gates[0], eta = gates[1], theta = gates[2];
    const float cg = 2.0f / ((float)BDIM * (float)VD);
    const size_t base = (size_t)BDIM * VD;
    float* outW = out + base + 1;
    float* outS = outW + (size_t)VD * KD;
    const float g  = cg * gradAcc[i];
    const float ns = eta * S[i] - theta * g;
    outS[i] = ns;
    outW[i] = (1.0f - alpha) * W[i] + ns;
    if (i == 0) out[base] = lossAcc[0] * (1.0f / ((float)BDIM * (float)VD));
}

// ---------------------------------------------------------------------------
extern "C" void kernel_launch(void* const* d_in, const int* in_sizes, int n_in,
                              void* d_out, int out_size, void* d_ws, size_t ws_size,
                              hipStream_t stream) {
    const float* k  = (const float*)d_in[0];
    const float* v  = (const float*)d_in[1];
    const float* W  = (const float*)d_in[2];
    const float* S  = (const float*)d_in[3];
    const float* gw = (const float*)d_in[4];
    const float* gb = (const float*)d_in[5];
    float* out = (float*)d_out;
    float* ws  = (float*)d_ws;

    // ws layout (floats): [0,512) ksum | [512,515) gates | [515] lossAcc | [1024, 1024+VD*KD) gradAcc
    float* ksum    = ws;
    float* gates   = ws + 512;
    float* lossAcc = ws + 515;
    float* gradAcc = ws + 1024;

    const int nzero = 1024 + VD * KD;
    zero_ws_kernel<<<(nzero + 255) / 256, 256, 0, stream>>>(ws, nzero);

    dim3 grid(BDIM / 128, VD / 64);
    titans_main_kernel<<<grid, 256, 0, stream>>>(k, v, W, out, gradAcc, lossAcc, ksum);

    gates_kernel<<<1, 32, 0, stream>>>(ksum, gw, gb, gates);
    epilogue_kernel<<<(VD * KD) / 256, 256, 0, stream>>>(W, S, gradAcc, gates, lossAcc, out);
}
